// ViSiLHead_22462678958548
// MI455X (gfx1250) — matrix operand order repaired and checked by reading, not compiled, not run on hardware
//
#include <hip/hip_runtime.h>

// ---------------------------------------------------------------------------
// Types for CDNA5 WMMA (wave32, 16x16x32 bf16 -> f32) and TDM descriptors
// ---------------------------------------------------------------------------
typedef __attribute__((ext_vector_type(16))) __bf16 v16bf;
typedef __attribute__((ext_vector_type(8)))  float  v8f;
typedef __attribute__((ext_vector_type(4)))  unsigned int v4u;
typedef __attribute__((ext_vector_type(8)))  int    v8i;
typedef __attribute__((ext_vector_type(4)))  int    v4i;
typedef unsigned short u16;
typedef unsigned int   u32;
typedef unsigned long long u64;

union alignas(32) Frag16 {
    v16bf v;
    uint4 q[2];
};

// ---------------------------------------------------------------------------
// Problem constants
// ---------------------------------------------------------------------------
#define TFR   512          // frames (T1 == T2)
#define RR    9            // regions per frame
#define KD    3840         // feature dim
#define GROWS (TFR * RR)   // 4608 GEMM rows
#define TILEF 16           // frames per block side
#define ROWS  (TILEF * RR) // 144 GEMM rows per block side
#define KTILE 64           // K elems staged per TDM round
#define LPAD  8            // extra u16 per LDS row (TDM pad: 4 DWORDs / 32 DWORDs)
#define LROW  (KTILE + LPAD) // 72 u16 = 144 B LDS row stride

__device__ inline u16 f2bf_rne(float f) {
    u32 u = __float_as_uint(f);
    u32 r = u + 0x7FFFu + ((u >> 16) & 1u);
    return (u16)(r >> 16);
}

// ---------------------------------------------------------------------------
// TDM: DMA a 144 x 64 bf16 tile (row stride KD elems) from global into LDS,
// inserting 4 DWORDs of padding every 32 DWORDs (-> 144B LDS row stride).
// D# layout per CDNA5 ISA ch.8 (group0 128b, group1 256b; groups 2/3 unused).
// This toolchain's builtin takes 6 args (g0, g1, g2, g3, g4, cpol).
// ---------------------------------------------------------------------------
__device__ inline void tdm_load_panel(const u16* gsrc, u32 lds_byte_off) {
    u64 ga = (u64)(uintptr_t)gsrc;
    v4u g0;
    g0[0] = 1u;                                  // count=1 (valid), no gather
    g0[1] = lds_byte_off;                        // LDS dest byte address
    g0[2] = (u32)ga;                             // global_addr[31:0]
    g0[3] = (u32)(ga >> 32) | (2u << 30);        // global_addr[56:32] | type=2
    v8i g1;
    //        data_size=2B     pad_enable      pad_interval=32dw pad_amount=4dw
    g1[0] = (int)((1u << 16) | (1u << 20) | (4u << 22) | (3u << 25));
    g1[1] = (int)((KD & 0xFFFFu) << 16);                       // tensor_dim0 lo
    g1[2] = (int)((KD >> 16) | ((GROWS & 0xFFFFu) << 16));     // dim0 hi | dim1 lo
    g1[3] = (int)((GROWS >> 16) | ((u32)KTILE << 16));         // dim1 hi | tile_dim0
    g1[4] = (int)ROWS;                                         // tile_dim1 (tile_dim2=0)
    g1[5] = (int)KD;                                           // tensor_dim0_stride lo
    g1[6] = 0;                                                 // stride hi | dim1_stride lo
    g1[7] = 0;
    v4i z4 = {0, 0, 0, 0};
    v8i z8 = {0, 0, 0, 0, 0, 0, 0, 0};
    __builtin_amdgcn_tensor_load_to_lds(g0, g1, z4, z4, z8, 0);
}

// ---------------------------------------------------------------------------
// 1) Convert q and t (f32, unit-normalized) to bf16
// ---------------------------------------------------------------------------
__global__ __launch_bounds__(256)
void cvt_bf16_kernel(const float* __restrict__ q, const float* __restrict__ t,
                     u16* __restrict__ qb, u16* __restrict__ tb, int n) {
    int idx = blockIdx.x * blockDim.x + threadIdx.x;
    if (idx >= n) return;
    qb[idx] = f2bf_rne(q[idx]);
    tb[idx] = f2bf_rne(t[idx]);
}

// ---------------------------------------------------------------------------
// 2) Tiled GEMM: C_block(144x144) = A_block(144xK) * B_block(144xK)^T
//    9 waves per block, wave w owns rows 16w..16w+15 (9 accumulator tiles).
//    B panel DMA'd into double-buffered LDS by the TDM; A read from global.
//    B fragments double-buffered in registers; a sched_barrier between the
//    load of fragment j+1 and the WMMA of fragment j pins the issue order so
//    the LDS loads overlap the previous WMMA (wait dscnt<=2 instead of 0).
// ---------------------------------------------------------------------------
__global__ __launch_bounds__(288)
void gemm_sim_kernel(const u16* __restrict__ qb, const u16* __restrict__ tb,
                     float* __restrict__ simC) {
    __shared__ u16 ldsB[2][ROWS * LROW];   // 2 x 20736 B

    const int bi   = blockIdx.x;
    const int bj   = blockIdx.y;
    const int tid  = threadIdx.x;
    const int wave = tid >> 5;           // 0..8
    const int lane = tid & 31;
    const int m    = lane & 15;          // row-in-tile (A) / col-in-tile (B,C)
    const int kh   = lane >> 4;          // lane half selects K sub-chunk

    const u16* Ab = qb + (size_t)(bi * ROWS) * KD;
    const u16* Bb = tb + (size_t)(bj * ROWS) * KD;

    v8f acc[9] = {};

    // Per-lane A source row (constant across K loop)
    const u16* arow = Ab + (size_t)(16 * wave + m) * KD + kh * 8;

    const int NR = KD / KTILE;           // 60 rounds
    const u32 lds0 = (u32)(uintptr_t)&ldsB[0][0];
    const u32 lds1 = (u32)(uintptr_t)&ldsB[1][0];

    if (tid < 32)
        tdm_load_panel(Bb, lds0);        // prologue: panel 0 in flight

    for (int r = 0; r < NR; ++r) {
        const int kk = r * KTILE;
        if (tid < 32) {
            if (r + 1 < NR) {
                tdm_load_panel(Bb + (kk + KTILE), ((r + 1) & 1) ? lds1 : lds0);
                __builtin_amdgcn_s_wait_tensorcnt(1);   // panel r landed
            } else {
                __builtin_amdgcn_s_wait_tensorcnt(0);
            }
        }
        __syncthreads();                 // publish panel r to all waves

        const u16* panel = &ldsB[r & 1][0];
        #pragma unroll
        for (int s = 0; s < 2; ++s) {    // two K=32 sub-steps per round
            // A fragment: 16-bit A 16x32 layout — half-lane 0: K 0-7 & 16-23,
            // half-lane 1: K 8-15 & 24-31 (two b128 loads)
            Frag16 a;
            a.q[0] = *(const uint4*)(arow + kk + s * 32);
            a.q[1] = *(const uint4*)(arow + kk + s * 32 + 16);

            // Explicit register double-buffer for B fragments
            Frag16 bf2[2];
            {
                const u16* brow = panel + (size_t)m * LROW + s * 32 + kh * 8;
                bf2[0].q[0] = *(const uint4*)(brow);
                bf2[0].q[1] = *(const uint4*)(brow + 16);
            }
            #pragma unroll
            for (int j = 0; j < 9; ++j) {
                if (j < 8) {
                    const u16* brow = panel + (size_t)((j + 1) * 16 + m) * LROW
                                            + s * 32 + kh * 8;
                    bf2[(j + 1) & 1].q[0] = *(const uint4*)(brow);
                    bf2[(j + 1) & 1].q[1] = *(const uint4*)(brow + 16);
                }
                // Pin issue order: loads for tile j+1 must stay ABOVE the
                // WMMA of tile j so they overlap its execution.
                __builtin_amdgcn_sched_barrier(0);
                acc[j] = __builtin_amdgcn_wmma_f32_16x16x32_bf16(
                    false, a.v, false, bf2[j & 1].v, (short)0, acc[j],
                    false, false);
            }
        }
        __syncthreads();                 // reads done before panel is reused
        if (r + 2 < NR)
            __builtin_prefetch(arow + kk + 2 * KTILE, 0, 1);  // global_prefetch
    }

    // Spill C tiles: lane holds column n=m, VGPR d -> row d + kh*8
    float* Cb = simC + (size_t)(bi * 32 + bj) * ROWS * ROWS;
    #pragma unroll
    for (int j = 0; j < 9; ++j) {
        #pragma unroll
        for (int d = 0; d < 8; ++d) {
            int r = 16 * wave + kh * 8 + d;
            int c = j * 16 + m;
            Cb[(size_t)r * ROWS + c] = acc[j][d];
        }
    }
}

// ---------------------------------------------------------------------------
// 3) f2f[i,j] = mean_o max_p C[il*9+o, jl*9+p]
// ---------------------------------------------------------------------------
__global__ __launch_bounds__(256)
void f2f_kernel(const float* __restrict__ simC, float* __restrict__ f2f) {
    int idx = blockIdx.x * blockDim.x + threadIdx.x;
    if (idx >= TFR * TFR) return;
    int i = idx >> 9, j = idx & 511;
    int bi = i >> 4, il = i & 15;
    int bj = j >> 4, jl = j & 15;
    const float* Cb = simC + (size_t)(bi * 32 + bj) * ROWS * ROWS;
    float s = 0.f;
    #pragma unroll
    for (int o = 0; o < 9; ++o) {
        const float* row = Cb + (size_t)(il * 9 + o) * ROWS + jl * 9;
        float mx = row[0];
        #pragma unroll
        for (int p = 1; p < 9; ++p) mx = fmaxf(mx, row[p]);
        s += mx;
    }
    f2f[idx] = s * (1.f / 9.f);
}

// ---------------------------------------------------------------------------
// 4) conv1 (1->32, 3x3 same) + relu + maxpool2 : 512^2 -> 32 x 256^2
// ---------------------------------------------------------------------------
__global__ __launch_bounds__(256)
void conv1_kernel(const float* __restrict__ in, const float* __restrict__ w1,
                  const float* __restrict__ b1, float* __restrict__ out) {
    int idx = blockIdx.x * blockDim.x + threadIdx.x;
    if (idx >= 32 * 256 * 256) return;
    int x = idx & 255, y = (idx >> 8) & 255, c = idx >> 16;
    const float* w = w1 + c * 9;
    float bias = b1[c];
    float mx = -3.4e38f;
    for (int dy = 0; dy < 2; ++dy)
        for (int dx = 0; dx < 2; ++dx) {
            int yy = 2 * y + dy, xx = 2 * x + dx;
            float acc = bias;
            #pragma unroll
            for (int ky = 0; ky < 3; ++ky)
                #pragma unroll
                for (int kx = 0; kx < 3; ++kx) {
                    int sy = yy + ky - 1, sx = xx + kx - 1;
                    if (sy >= 0 && sy < 512 && sx >= 0 && sx < 512)
                        acc += w[ky * 3 + kx] * in[sy * 512 + sx];
                }
            mx = fmaxf(mx, fmaxf(acc, 0.f));
        }
    out[idx] = mx;
}

// ---------------------------------------------------------------------------
// 5) conv2 (32->64, 3x3 same) + relu + maxpool2 : 32x256^2 -> 64x128^2
// ---------------------------------------------------------------------------
__global__ __launch_bounds__(256)
void conv2_kernel(const float* __restrict__ in, const float* __restrict__ w2,
                  const float* __restrict__ b2, float* __restrict__ out) {
    int idx = blockIdx.x * blockDim.x + threadIdx.x;
    if (idx >= 64 * 128 * 128) return;
    int x = idx & 127, y = (idx >> 7) & 127, c = idx >> 14;
    const float* w = w2 + c * 32 * 9;
    float bias = b2[c];
    float mx = -3.4e38f;
    for (int dy = 0; dy < 2; ++dy)
        for (int dx = 0; dx < 2; ++dx) {
            int yy = 2 * y + dy, xx = 2 * x + dx;
            float acc = bias;
            for (int ci = 0; ci < 32; ++ci) {
                const float* wp = w + ci * 9;
                const float* ip = in + ci * 256 * 256;
                #pragma unroll
                for (int ky = 0; ky < 3; ++ky)
                    #pragma unroll
                    for (int kx = 0; kx < 3; ++kx) {
                        int sy = yy + ky - 1, sx = xx + kx - 1;
                        if (sy >= 0 && sy < 256 && sx >= 0 && sx < 256)
                            acc += wp[ky * 3 + kx] * ip[sy * 256 + sx];
                    }
            }
            mx = fmaxf(mx, fmaxf(acc, 0.f));
        }
    out[idx] = mx;
}

// ---------------------------------------------------------------------------
// 6) conv3 (64->128, 3x3 same) + relu : 64x128^2 -> 128x128^2
// ---------------------------------------------------------------------------
__global__ __launch_bounds__(256)
void conv3_kernel(const float* __restrict__ in, const float* __restrict__ w3,
                  const float* __restrict__ b3, float* __restrict__ out) {
    int idx = blockIdx.x * blockDim.x + threadIdx.x;
    if (idx >= 128 * 128 * 128) return;
    int x = idx & 127, y = (idx >> 7) & 127, c = idx >> 14;
    const float* w = w3 + c * 64 * 9;
    float acc = b3[c];
    for (int ci = 0; ci < 64; ++ci) {
        const float* wp = w + ci * 9;
        const float* ip = in + ci * 128 * 128;
        #pragma unroll
        for (int ky = 0; ky < 3; ++ky)
            #pragma unroll
            for (int kx = 0; kx < 3; ++kx) {
                int sy = y + ky - 1, sx = x + kx - 1;
                if (sy >= 0 && sy < 128 && sx >= 0 && sx < 128)
                    acc += wp[ky * 3 + kx] * ip[sy * 128 + sx];
            }
    }
    out[idx] = fmaxf(acc, 0.f);
}

// ---------------------------------------------------------------------------
// 7) 1x1 conv (128->1) + bias + clip [-1,1]
// ---------------------------------------------------------------------------
__global__ __launch_bounds__(256)
void final_map_kernel(const float* __restrict__ in, const float* __restrict__ wf,
                      const float* __restrict__ bf, float* __restrict__ scores) {
    int idx = blockIdx.x * blockDim.x + threadIdx.x;
    if (idx >= 128 * 128) return;
    float acc = bf[0];
    for (int c = 0; c < 128; ++c)
        acc += wf[c] * in[c * 128 * 128 + idx];
    scores[idx] = fminf(fmaxf(acc, -1.f), 1.f);
}

// ---------------------------------------------------------------------------
// 8) out = mean_y max_x scores[y][x]
// ---------------------------------------------------------------------------
__global__ __launch_bounds__(128)
void final_reduce_kernel(const float* __restrict__ scores, float* __restrict__ out) {
    __shared__ float red[128];
    int y = threadIdx.x;
    float mx = scores[y * 128];
    for (int x = 1; x < 128; ++x) mx = fmaxf(mx, scores[y * 128 + x]);
    red[y] = mx;
    __syncthreads();
    for (int s = 64; s > 0; s >>= 1) {
        if (y < s) red[y] += red[y + s];
        __syncthreads();
    }
    if (y == 0) out[0] = red[0] * (1.f / 128.f);
}

// ---------------------------------------------------------------------------
// Host launcher
// ---------------------------------------------------------------------------
extern "C" void kernel_launch(void* const* d_in, const int* in_sizes, int n_in,
                              void* d_out, int out_size, void* d_ws, size_t ws_size,
                              hipStream_t stream) {
    const float* q  = (const float*)d_in[0];
    const float* t  = (const float*)d_in[1];
    const float* w1 = (const float*)d_in[2];
    const float* b1 = (const float*)d_in[3];
    const float* w2 = (const float*)d_in[4];
    const float* b2 = (const float*)d_in[5];
    const float* w3 = (const float*)d_in[6];
    const float* b3 = (const float*)d_in[7];
    const float* wf = (const float*)d_in[8];
    const float* bf = (const float*)d_in[9];
    float* out = (float*)d_out;

    char* ws = (char*)d_ws;
    const size_t nQT = (size_t)GROWS * KD;                 // 17,694,720 elems
    u16*   qb   = (u16*)(ws);                              // 35.4 MB
    u16*   tb   = (u16*)(ws + nQT * 2);                    // 35.4 MB
    float* simC = (float*)(ws + nQT * 4);                  // 84.9 MB
    float* f2f  = (float*)(ws + nQT * 4 + (size_t)1024 * ROWS * ROWS * 4); // 1 MB
    // Conv buffers reuse the qb/tb region (free after the GEMM pass)
    float* c1     = (float*)(ws);                          //  8.4 MB
    float* c2     = (float*)(ws + (size_t)32 * 256 * 256 * 4);
    float* c3     = (float*)(ws + (size_t)32 * 256 * 256 * 4 + (size_t)64 * 128 * 128 * 4);
    float* scores = (float*)(ws + (size_t)32 * 256 * 256 * 4 + (size_t)64 * 128 * 128 * 4
                                 + (size_t)128 * 128 * 128 * 4);

    // 1) f32 -> bf16
    {
        int n = (int)nQT;
        cvt_bf16_kernel<<<(n + 255) / 256, 256, 0, stream>>>(q, t, qb, tb, n);
    }
    // 2) WMMA GEMM: 32x32 blocks of 144x144 output, TDM-staged B panels
    gemm_sim_kernel<<<dim3(32, 32), 288, 0, stream>>>(qb, tb, simC);
    // 3) mean-of-max reduction to the 512x512 similarity map
    f2f_kernel<<<(TFR * TFR + 255) / 256, 256, 0, stream>>>(simC, f2f);
    // 4-6) conv stack
    conv1_kernel<<<(32 * 256 * 256 + 255) / 256, 256, 0, stream>>>(f2f, w1, b1, c1);
    conv2_kernel<<<(64 * 128 * 128 + 255) / 256, 256, 0, stream>>>(c1, w2, b2, c2);
    conv3_kernel<<<(128 * 128 * 128 + 255) / 256, 256, 0, stream>>>(c2, w3, b3, c3);
    // 7) 1x1 conv + clip
    final_map_kernel<<<(128 * 128 + 255) / 256, 256, 0, stream>>>(c3, wf, bf, scores);
    // 8) final scalar
    final_reduce_kernel<<<1, 128, 0, stream>>>(scores, out);
}
